// SoftmaxLoss_86096914415969
// MI455X (gfx1250) — matrix operand order repaired and checked by reading, not compiled, not run on hardware
//
#include <hip/hip_runtime.h>

// ---------------------------------------------------------------------------
// Problem constants (match reference)
// ---------------------------------------------------------------------------
#define ALPHA_C   10.0f
#define BETA_C    2.0f
#define NCLS      98
#define EMB       512
#define NROWS     8192

// CDNA5 WMMA vector types
typedef _Float16 half8  __attribute__((ext_vector_type(8)));
typedef _Float16 v16h_t __attribute__((ext_vector_type(16)));
typedef float    v8f_t  __attribute__((ext_vector_type(8)));

// ---------------------------------------------------------------------------
// Zero a float region (workspace is poisoned 0xAA; must re-zero every call)
// ---------------------------------------------------------------------------
__global__ void sml_zero_kernel(float* __restrict__ p, int n) {
    for (int i = blockIdx.x * blockDim.x + threadIdx.x; i < n;
         i += gridDim.x * blockDim.x)
        p[i] = 0.0f;
}

// ---------------------------------------------------------------------------
// f32 -> f16 copy of the embeddings (8 MB; lives in L2 for the big GEMM)
// ---------------------------------------------------------------------------
__global__ void sml_cvt_kernel(const float* __restrict__ X,
                               _Float16* __restrict__ X16) {
    const size_t total = (size_t)NROWS * EMB;
    for (size_t i = blockIdx.x * (size_t)blockDim.x + threadIdx.x; i < total;
         i += (size_t)gridDim.x * blockDim.x)
        X16[i] = (_Float16)X[i];
}

// ---------------------------------------------------------------------------
// L2-normalize kernel columns: Wn[:,c] = K[:,c] / ||K[:,c]||  (one block/col)
// ---------------------------------------------------------------------------
__global__ void sml_colnorm_kernel(const float* __restrict__ K,
                                   float* __restrict__ Wn) {
    __shared__ float red[64];
    const int c = blockIdx.x;
    const int t = threadIdx.x;
    float s = 0.0f;
    for (int k = t; k < EMB; k += 64) {
        float v = K[k * NCLS + c];
        s += v * v;
    }
    red[t] = s;
    __syncthreads();
    for (int off = 32; off >= 1; off >>= 1) {
        if (t < off) red[t] += red[t + off];
        __syncthreads();
    }
    const float rn = rsqrtf(red[0]);
    for (int k = t; k < EMB; k += 64)
        Wn[k * NCLS + c] = K[k * NCLS + c] * rn;
}

// ---------------------------------------------------------------------------
// Class histogram (used for analytic negative counts)
// ---------------------------------------------------------------------------
__global__ void sml_hist_kernel(const int* __restrict__ tgt,
                                int* __restrict__ hist) {
    const int i = blockIdx.x * blockDim.x + threadIdx.x;
    if (i < NROWS) atomicAdd(&hist[tgt[i]], 1);
}

// ---------------------------------------------------------------------------
// Margin-softmax cross entropy + precision. One wave32 per row,
// 4 classes per lane (98 classes), wave-shuffle reductions.
// ---------------------------------------------------------------------------
__global__ void __launch_bounds__(256)
sml_ce_kernel(const float* __restrict__ X, const int* __restrict__ tgt,
              const float* __restrict__ Wn, float* __restrict__ lossSum,
              float* __restrict__ correctSum) {
    __shared__ float xs[8][EMB];
    const int lane = threadIdx.x & 31;
    const int wave = threadIdx.x >> 5;
    const int row  = blockIdx.x * 8 + wave;

    const float* xr = X + (size_t)row * EMB;
    for (int k = lane; k < EMB; k += 32) xs[wave][k] = xr[k];
    __syncthreads();

    const int t = tgt[row];

    float acc4[4] = {0.f, 0.f, 0.f, 0.f};
    for (int k = 0; k < EMB; ++k) {
        const float xk = xs[wave][k];
        const float* wrow = Wn + k * NCLS;
#pragma unroll
        for (int q = 0; q < 4; ++q) {
            const int c = lane + 32 * q;
            if (c < NCLS) acc4[q] += xk * wrow[c];
        }
    }

    float lg[4];
    float bv = -1e30f;
    int   bi = 0x7fffffff;
#pragma unroll
    for (int q = 0; q < 4; ++q) {
        const int c = lane + 32 * q;
        if (c < NCLS) {
            float cosv = fminf(fmaxf(acc4[q], -1.0f), 1.0f);
            float l = (cosv - ((c == t) ? BETA_C : 0.0f)) * ALPHA_C;
            lg[q] = l;
            if (l > bv || (l == bv && c < bi)) { bv = l; bi = c; }
        } else {
            lg[q] = -1e30f;
        }
    }
    // wave max + argmax (first index wins on ties)
#pragma unroll
    for (int msk = 16; msk >= 1; msk >>= 1) {
        float ov = __shfl_xor(bv, msk, 32);
        int   oi = __shfl_xor(bi, msk, 32);
        if (ov > bv || (ov == bv && oi < bi)) { bv = ov; bi = oi; }
    }
    float se = 0.0f, tl = 0.0f;
#pragma unroll
    for (int q = 0; q < 4; ++q) {
        const int c = lane + 32 * q;
        if (c < NCLS) {
            se += __expf(lg[q] - bv);
            if (c == t) tl = lg[q];
        }
    }
#pragma unroll
    for (int msk = 16; msk >= 1; msk >>= 1) {
        se += __shfl_xor(se, msk, 32);
        tl += __shfl_xor(tl, msk, 32);
    }
    if (lane == 0) {
        const float loss = (bv + __logf(se)) - tl;
        atomicAdd(lossSum, loss);
        atomicAdd(correctSum, (bi == t) ? 1.0f : 0.0f);
    }
}

// ---------------------------------------------------------------------------
// Fused sim = X X^T with masked per-row pos/neg accumulation.
// Grid (64, 4): block owns 128 rows (staged in 128 KB dynamic LDS) x 2048 cols.
// Each wave32 owns 16 rows and processes TWO 16-col tiles per pass with two
// independent accumulator chains sharing the register-resident A fragments:
// while one chain waits on its B loads, the other chain's WMMA issues.
// The next tile-pair's full 2x16x1KB B panel is prefetched (8 lines per row,
// split across the two half-waves) so B loads hit near caches.
//
// Fragment addressing used here (contiguous k order, same for A and B):
//  A (16x32 f16): lane L -> row M=L&15; holds 16 consecutive K starting at
//                 16*(L>>4)  (two 16B LDS reads)
//  B (32x16 f16): lane L -> col N=L&15; holds 16 consecutive K starting at
//                 16*(L>>4)  (two 16B global reads from the f16 row of col N)
//  C (16x16 f32): lane L -> col N=L&15; VGPR r -> row r + 8*(L>>4)
// ---------------------------------------------------------------------------
#define SIM_ROWS_PER_BLOCK 128
#define SIM_COL_CHUNK      (NROWS / 4)   // 2048
#define KSTEPS             (EMB / 32)    // 16 WMMA per tile

__global__ void __launch_bounds__(256)
sml_sim_kernel(const _Float16* __restrict__ X16, const int* __restrict__ tgt,
               float* __restrict__ rowPos, float* __restrict__ rowNeg,
               float* __restrict__ rowCnt) {
    extern __shared__ char smem[];
    _Float16* ldsX  = (_Float16*)smem;                                  // 128x512 f16
    int*      ldsCl = (int*)(smem + SIM_ROWS_PER_BLOCK * EMB * sizeof(_Float16));

    const int rowBase  = blockIdx.x * SIM_ROWS_PER_BLOCK;
    const int colChunk = blockIdx.y * SIM_COL_CHUNK;

    // ---- cooperative stage of this block's 128 rows into LDS (b128 chunks)
    {
        const uint4* src = (const uint4*)(X16 + (size_t)rowBase * EMB);
        uint4* dst = (uint4*)ldsX;
        for (int i = threadIdx.x; i < SIM_ROWS_PER_BLOCK * EMB / 8; i += 256)
            dst[i] = src[i];
        if (threadIdx.x < SIM_ROWS_PER_BLOCK)
            ldsCl[threadIdx.x] = tgt[rowBase + threadIdx.x];
    }
    __syncthreads();

    const int lane = threadIdx.x & 31;
    const int wave = threadIdx.x >> 5;
    const int m    = lane & 15;   // A row within tile / C column
    const int hf   = lane >> 4;   // half-wave select

    const int waveRow0 = wave * 16;   // local row base for this wave

    float pS[8], nS[8], pC[8];
    int   rCls[8];
    const int rId0 = rowBase + waveRow0 + 8 * hf;   // C row for slot r is rId0+r
#pragma unroll
    for (int r = 0; r < 8; ++r) {
        pS[r] = 0.0f; nS[r] = 0.0f; pC[r] = 0.0f;
        rCls[r] = ldsCl[waveRow0 + r + 8 * hf];
    }

    const _Float16* aBase = ldsX + (size_t)(waveRow0 + m) * EMB;

    for (int ct = 0; ct < SIM_COL_CHUNK / 32; ++ct) {
        const int colBase = colChunk + ct * 32;
        const int col0    = colBase + m;         // tile 0 column for this lane
        const int col1    = col0 + 16;           // tile 1 column for this lane
        const int cCls0   = tgt[col0];
        const int cCls1   = tgt[col1];
        const half8* bPtr0 = (const half8*)(X16 + (size_t)col0 * EMB + 16 * hf);
        const half8* bPtr1 = (const half8*)(X16 + (size_t)col1 * EMB + 16 * hf);

        // prefetch the whole next tile-pair B panel (each row is 1 KB f16;
        // half-waves cover the two 512 B halves, 4 lines each at +128 B)
        if (ct + 1 < SIM_COL_CHUNK / 32) {
            const _Float16* p0 = X16 + (size_t)(col0 + 32) * EMB + hf * 256;
            const _Float16* p1 = X16 + (size_t)(col1 + 32) * EMB + hf * 256;
#pragma unroll
            for (int q = 0; q < 4; ++q) {
                __builtin_prefetch(p0 + q * 64, 0, 3);
                __builtin_prefetch(p1 + q * 64, 0, 3);
            }
        }

        v8f_t acc0 = {};
        v8f_t acc1 = {};
#pragma unroll
        for (int kt = 0; kt < KSTEPS; ++kt) {
            union { v16h_t v; half8 h[2]; } A, B0, B1;
            // issue both chains' loads before either WMMA: while chain 0
            // waits, chain 1's loads are in flight (and vice versa)
            B0.h[0] = bPtr0[4 * kt];
            B0.h[1] = bPtr0[4 * kt + 1];
            B1.h[0] = bPtr1[4 * kt];
            B1.h[1] = bPtr1[4 * kt + 1];
            const int kk = 32 * kt;
            A.h[0] = *(const half8*)(aBase + kk + 16 * hf);
            A.h[1] = *(const half8*)(aBase + kk + 16 * hf + 8);
            acc0 = __builtin_amdgcn_wmma_f32_16x16x32_f16(
                false, A.v, false, B0.v, (short)0, acc0, false, false);
            acc1 = __builtin_amdgcn_wmma_f32_16x16x32_f16(
                false, A.v, false, B1.v, (short)0, acc1, false, false);
        }

        // masked accumulation directly from the two C fragments
#pragma unroll
        for (int r = 0; r < 8; ++r) {
            const int rid = rId0 + r;
            {
                const float s   = acc0[r];
                const bool same = (rCls[r] == cCls0);
                const bool pos  = same && (rid != col0) && (s < 1.0f);
                pS[r] += pos ? s : 0.0f;
                pC[r] += pos ? 1.0f : 0.0f;
                nS[r] += same ? 0.0f : s;
            }
            {
                const float s   = acc1[r];
                const bool same = (rCls[r] == cCls1);
                const bool pos  = same && (rid != col1) && (s < 1.0f);
                pS[r] += pos ? s : 0.0f;
                pC[r] += pos ? 1.0f : 0.0f;
                nS[r] += same ? 0.0f : s;
            }
        }
    }

    // reduce each row across its 16 lanes (halves are independent: width=16)
#pragma unroll
    for (int r = 0; r < 8; ++r) {
        float a = pS[r], b = nS[r], c = pC[r];
#pragma unroll
        for (int msk = 8; msk >= 1; msk >>= 1) {
            a += __shfl_xor(a, msk, 16);
            b += __shfl_xor(b, msk, 16);
            c += __shfl_xor(c, msk, 16);
        }
        if (m == 0) {
            atomicAdd(&rowPos[rId0 + r], a);
            atomicAdd(&rowNeg[rId0 + r], b);
            atomicAdd(&rowCnt[rId0 + r], c);
        }
    }
}

// ---------------------------------------------------------------------------
// Finalize: four scalars -> d_out = {loss, precision, mean_pos, mean_neg}
// ---------------------------------------------------------------------------
__global__ void __launch_bounds__(256)
sml_fin_kernel(const float* __restrict__ rowPos, const float* __restrict__ rowNeg,
               const float* __restrict__ rowCnt, const int* __restrict__ tgt,
               const int* __restrict__ hist, const float* __restrict__ lossSum,
               const float* __restrict__ correctSum, float* __restrict__ out) {
    __shared__ float rp[256], rn[256];
    float ps = 0.0f, ns = 0.0f;
    for (int i = threadIdx.x; i < NROWS; i += 256) {
        ps += rowPos[i] / rowCnt[i];
        ns += rowNeg[i] / (float)(NROWS - hist[tgt[i]]);
    }
    rp[threadIdx.x] = ps;
    rn[threadIdx.x] = ns;
    __syncthreads();
    for (int off = 128; off >= 1; off >>= 1) {
        if (threadIdx.x < off) {
            rp[threadIdx.x] += rp[threadIdx.x + off];
            rn[threadIdx.x] += rn[threadIdx.x + off];
        }
        __syncthreads();
    }
    if (threadIdx.x == 0) {
        out[0] = lossSum[0] / (float)NROWS;
        out[1] = correctSum[0] / (float)NROWS;
        out[2] = rp[0] / (float)NROWS;
        out[3] = rn[0] / (float)NROWS;
    }
}

// ---------------------------------------------------------------------------
// Host launcher (graph-capture safe: only kernel launches on `stream`)
// ---------------------------------------------------------------------------
extern "C" void kernel_launch(void* const* d_in, const int* in_sizes, int n_in,
                              void* d_out, int out_size, void* d_ws, size_t ws_size,
                              hipStream_t stream) {
    (void)in_sizes; (void)n_in; (void)out_size; (void)ws_size;

    const float* X   = (const float*)d_in[0];   // [8192, 512] f32 (unit rows)
    const int*   tgt = (const int*)d_in[1];     // [8192] class ids
    const float* Ker = (const float*)d_in[2];   // [512, 98] f32

    char* ws = (char*)d_ws;
    size_t off = 0;
    _Float16* X16 = (_Float16*)(ws + off); off += (size_t)NROWS * EMB * sizeof(_Float16);
    float* Wn     = (float*)(ws + off);    off += (size_t)EMB * NCLS * sizeof(float);
    off = (off + 255) & ~(size_t)255;
    float* rowPos = (float*)(ws + off);    off += (size_t)NROWS * sizeof(float);
    float* rowNeg = (float*)(ws + off);    off += (size_t)NROWS * sizeof(float);
    float* rowCnt = (float*)(ws + off);    off += (size_t)NROWS * sizeof(float);
    int*   hist   = (int*)(ws + off);      off += 128 * sizeof(int);
    float* lossSum    = (float*)(ws + off); off += sizeof(float);
    float* correctSum = (float*)(ws + off); off += sizeof(float);

    // rowPos .. correctSum are contiguous: zero them in one pass
    const int zeroFloats = (int)(((ws + off) - (char*)rowPos) / sizeof(float));
    sml_zero_kernel<<<64, 256, 0, stream>>>(rowPos, zeroFloats);

    sml_cvt_kernel<<<2048, 256, 0, stream>>>(X, X16);
    sml_colnorm_kernel<<<NCLS, 64, 0, stream>>>(Ker, Wn);
    sml_hist_kernel<<<(NROWS + 255) / 256, 256, 0, stream>>>(tgt, hist);

    sml_ce_kernel<<<NROWS / 8, 256, 0, stream>>>(X, tgt, Wn, lossSum, correctSum);

    dim3 simGrid(NROWS / SIM_ROWS_PER_BLOCK, NROWS / SIM_COL_CHUNK);  // (64, 4)
    const size_t simShmem = (size_t)SIM_ROWS_PER_BLOCK * EMB * sizeof(_Float16)
                          + SIM_ROWS_PER_BLOCK * sizeof(int);          // 128 KB + 512 B
    sml_sim_kernel<<<simGrid, 256, simShmem, stream>>>(X16, tgt, rowPos, rowNeg, rowCnt);

    sml_fin_kernel<<<1, 256, 0, stream>>>(rowPos, rowNeg, rowCnt, tgt, hist,
                                          lossSum, correctSum, (float*)d_out);
}
